// LinearStateLayer_37014028157376
// MI455X (gfx1250) — compile-verified, matrix operand order checked by reading, not run on hardware
//
#include <hip/hip_runtime.h>

// ---------------- problem constants ----------------
#define Bb 4
#define Tt 4096
#define Hh 1024
#define Ss 64
#define Mm (Bb * Tt)   // 16384 rows
#define NC 64          // scan chunks
#define TC (Tt / NC)   // 64 steps per chunk
#define LN_EPS 1e-5f

typedef __attribute__((ext_vector_type(16))) __bf16 v16bf;
typedef __attribute__((ext_vector_type(8)))  float  v8f;
typedef __attribute__((ext_vector_type(4)))  int    vi4;
typedef __attribute__((ext_vector_type(2)))  int    vi2;

// ---------------- CDNA5 async global->LDS staging ----------------
#if __has_builtin(__builtin_amdgcn_global_load_async_to_lds_b128) && \
    __has_builtin(__builtin_amdgcn_global_load_async_to_lds_b64)
#define HAVE_ASYNC 1
#else
#define HAVE_ASYNC 0
#endif

typedef __attribute__((address_space(1))) vi4* g4p_t;   // global int4*
typedef __attribute__((address_space(3))) vi4* l4p_t;   // shared int4*
typedef __attribute__((address_space(1))) vi2* g2p_t;   // global int2*
typedef __attribute__((address_space(3))) vi2* l2p_t;   // shared int2*

static __device__ inline void stage16B(__bf16* ldst, const __bf16* gsrc) {
#if HAVE_ASYNC
  __builtin_amdgcn_global_load_async_to_lds_b128((g4p_t)gsrc, (l4p_t)ldst, 0, 0);
#else
  *(uint4*)ldst = *(const uint4*)gsrc;
#endif
}

static __device__ inline void stage8B(__bf16* ldst, const __bf16* gsrc) {
#if HAVE_ASYNC
  __builtin_amdgcn_global_load_async_to_lds_b64((g2p_t)gsrc, (l2p_t)ldst, 0, 0);
#else
  *(uint2*)ldst = *(const uint2*)gsrc;
#endif
}

template <int N>
static __device__ inline void wait_async() {
#if HAVE_ASYNC
#if __has_builtin(__builtin_amdgcn_s_wait_asynccnt)
  __builtin_amdgcn_s_wait_asynccnt(N);
#else
  asm volatile("s_wait_asynccnt %0" ::"n"(N) : "memory");
#endif
#endif
}

// D = A(16x32 bf16) x B(32x16 bf16) + C, f32 accumulate, wave32 WMMA
static __device__ inline v8f wmma_bf16(v16bf a, v16bf b, v8f c) {
  return __builtin_amdgcn_wmma_f32_16x16x32_bf16(
      /*neg_a=*/false, a, /*neg_b=*/false, b,
      /*c_mod=*/(short)0, c, /*reuse_a=*/false, /*reuse_b=*/false);
}

// A fragment from LDS tile stored row-major [rows][ldk] bf16.
static __device__ inline v16bf frag_a_lds(const __bf16* lds, int ldk, int lane,
                                          int rowBase, int ks) {
  return *(const v16bf*)(lds + (size_t)(rowBase + (lane & 15)) * ldk + ks +
                         ((lane >> 4) << 4));
}

// B fragment straight from a row-major [N][K] bf16 weight matrix
// (weights are tiny and L2-resident; 32B contiguous per lane).
static __device__ inline v16bf frag_b_glb(const __bf16* w, int ldk, int n0, int kk, int lane) {
  return *(const v16bf*)(w + (size_t)(n0 + (lane & 15)) * ldk + kk + ((lane >> 4) << 4));
}

// ---------------- fp32 -> bf16 convert ----------------
__global__ __launch_bounds__(256) void k_cvt_bf16(const float* __restrict__ src,
                                                  __bf16* __restrict__ dst, int n) {
  int i = blockIdx.x * 256 + threadIdx.x;
  if (i < n) dst[i] = (__bf16)src[i];
}

// ---------------- K1: u = sigmoid(x Wg^T + bg) * (x WB^T) ----------------
// M=16384, N=64, K=1024. Block = 256 thr (8 waves = 2 row-tiles x 4 col-tiles),
// block tile 32 rows x 64 cols. A double-buffer staged to LDS asynchronously
// in 32x64 panels (4 KB); each stage feeds 2 k-steps x 2 matrices = 4 WMMAs.
__global__ __launch_bounds__(256) void k_gemm_u(const __bf16* __restrict__ xb,
                                                const __bf16* __restrict__ Wg_bf,
                                                const __bf16* __restrict__ WB_bf,
                                                const float* __restrict__ bg,
                                                float* __restrict__ u) {
  __shared__ __bf16 ldsA[2][32 * 64];
  const int tid = threadIdx.x, lane = tid & 31, w = tid >> 5;
  const int wrow = w >> 2, wcol = w & 3;
  const int rowBlk = blockIdx.x * 32;
  const int n0 = wcol * 16;
  const int sr = tid >> 3, sc = (tid & 7) * 8;          // 32 rows x 8 chunks of 16B
  const __bf16* gA = xb + (size_t)(rowBlk + sr) * Hh + sc;
  const int lofs = sr * 64 + sc;

  stage16B(&ldsA[0][lofs], gA);                          // prologue: kk=0 -> buf0
  v8f accg = {}, accb = {};
  const int NI = Hh / 64;                                // 16 stages
  for (int i = 0; i < NI; ++i) {
    const int kk = i * 64;
    const int p = i & 1;
    if (i + 1 < NI) {
      stage16B(&ldsA[p ^ 1][lofs], gA + kk + 64);        // overlap next stage
      wait_async<1>();
    } else {
      wait_async<0>();
    }
    __syncthreads();
    const __bf16* bufA = &ldsA[p][0];
    for (int ks = 0; ks < 64; ks += 32) {
      v16bf a  = frag_a_lds(bufA, 64, lane, wrow * 16, ks);
      v16bf b1 = frag_b_glb(Wg_bf, Hh, n0, kk + ks, lane);
      v16bf b2 = frag_b_glb(WB_bf, Hh, n0, kk + ks, lane);
      accg = wmma_bf16(a, b1, accg);
      accb = wmma_bf16(a, b2, accb);
    }
    __syncthreads();
  }
  const int col = n0 + (lane & 15);
  const int rowBase = rowBlk + wrow * 16 + ((lane >> 4) << 3);
  const float bgv = bg[col];
  for (int i = 0; i < 8; ++i) {
    float z = accg[i] + bgv;
    float g = 1.0f / (1.0f + __expf(-z));
    u[(size_t)(rowBase + i) * Ss + col] = g * accb[i];
  }
}

// ---------------- scan pass 1: per-chunk carries (zero-init local scan) ----------------
__global__ __launch_bounds__(64) void k_scan_carry(const float* __restrict__ u,
                                                   const float* __restrict__ A_log,
                                                   float* __restrict__ carry) {
  const int s = threadIdx.x;
  const int b = blockIdx.x / NC, ch = blockIdx.x % NC;
  const float a = __expf(A_log[s]);
  const float* up = u + ((size_t)b * Tt + ch * TC) * Ss + s;
  float c = 0.f;
  for (int i = 0; i < TC; ++i) c = fmaf(a, c, up[(size_t)i * Ss]);
  carry[((size_t)b * NC + ch) * Ss + s] = c;
}

// ---------------- scan pass 2: sequential combine over 64 chunks ----------------
__global__ __launch_bounds__(256) void k_scan_combine(const float* __restrict__ carry,
                                                      const float* __restrict__ A_log,
                                                      float* __restrict__ start,
                                                      float* __restrict__ fs_out) {
  const int tid = threadIdx.x;       // 256 = (b,s)
  const int b = tid >> 6, s = tid & 63;
  const float a = __expf(A_log[s]);
  float aL = a;                      // a^TC via 6 squarings (TC=64)
  for (int k = 0; k < 6; ++k) aL *= aL;
  float st = 0.f;
  for (int ch = 0; ch < NC; ++ch) {
    start[((size_t)b * NC + ch) * Ss + s] = st;
    st = fmaf(aL, st, carry[((size_t)b * NC + ch) * Ss + s]);
  }
  fs_out[b * Ss + s] = st;           // final_state output
}

// ---------------- scan pass 3: replay chunks with correct init, emit bf16 states ----
__global__ __launch_bounds__(64) void k_scan_states(const float* __restrict__ u,
                                                    const float* __restrict__ A_log,
                                                    const float* __restrict__ start,
                                                    __bf16* __restrict__ st_bf) {
  const int s = threadIdx.x;
  const int b = blockIdx.x / NC, ch = blockIdx.x % NC;
  const float a = __expf(A_log[s]);
  float st = start[((size_t)b * NC + ch) * Ss + s];
  const float* up = u + ((size_t)b * Tt + ch * TC) * Ss + s;
  __bf16* op = st_bf + ((size_t)b * Tt + ch * TC) * Ss + s;
  for (int i = 0; i < TC; ++i) {
    st = fmaf(a, st, up[(size_t)i * Ss]);
    op[(size_t)i * Ss] = (__bf16)st;
  }
}

// ---------------- K5: y = states WC^T + D*x + x, LayerNorm -> hn (bf16) ----------
// M=16384, K=64, N=1024. One block owns 16 full rows (needed for LN).
// 8 waves x 8 col-tiles = 1024 cols. LN stats via LDS float atomics.
__global__ __launch_bounds__(256) void k_gemm_y_ln(const __bf16* __restrict__ st_bf,
                                                   const __bf16* __restrict__ WC_bf,
                                                   const float* __restrict__ x,
                                                   const float* __restrict__ Dv,
                                                   const float* __restrict__ gamma,
                                                   const float* __restrict__ beta,
                                                   __bf16* __restrict__ hn_bf) {
  __shared__ __bf16 ldsA[16 * 64];
  __shared__ float ldsSum[16], ldsSq[16], ldsMu[16], ldsRs[16];
  const int tid = threadIdx.x, lane = tid & 31, w = tid >> 5;
  const int row0 = blockIdx.x * 16;
  {
    const int r = tid >> 4, c4 = (tid & 15) * 4;
    stage8B(&ldsA[r * 64 + c4], st_bf + (size_t)(row0 + r) * Ss + c4);
  }
  if (tid < 16) { ldsSum[tid] = 0.f; ldsSq[tid] = 0.f; }
  wait_async<0>();
  __syncthreads();

  v8f zero = {};
  v8f acc[8];
  for (int j = 0; j < 8; ++j) acc[j] = zero;
  for (int kk = 0; kk < Ss; kk += 32) {
    v16bf a = frag_a_lds(ldsA, 64, lane, 0, kk);
    for (int j = 0; j < 8; ++j) {
      const int n0 = w * 128 + j * 16;
      v16bf bfr = frag_b_glb(WC_bf, Ss, n0, kk, lane);
      acc[j] = wmma_bf16(a, bfr, acc[j]);
    }
  }
  // h = y + D*x + x; accumulate per-row LN statistics
  const int rloc = (lane >> 4) << 3;           // 0 or 8
  const int rowB = row0 + rloc;
  float s_[8] = {0, 0, 0, 0, 0, 0, 0, 0};
  float q_[8] = {0, 0, 0, 0, 0, 0, 0, 0};
  for (int j = 0; j < 8; ++j) {
    const int col = w * 128 + j * 16 + (lane & 15);
    const float d = Dv[col];
    for (int i = 0; i < 8; ++i) {
      float xv = x[(size_t)(rowB + i) * Hh + col];
      float h = acc[j][i] + d * xv + xv;
      acc[j][i] = h;
      s_[i] += h;
      q_[i] += h * h;
    }
  }
  for (int i = 0; i < 8; ++i) {
    atomicAdd(&ldsSum[rloc + i], s_[i]);
    atomicAdd(&ldsSq[rloc + i], q_[i]);
  }
  __syncthreads();
  if (tid < 16) {
    float mu = ldsSum[tid] * (1.0f / (float)Hh);
    float var = ldsSq[tid] * (1.0f / (float)Hh) - mu * mu;
    ldsMu[tid] = mu;
    ldsRs[tid] = rsqrtf(var + LN_EPS);
  }
  __syncthreads();
  for (int j = 0; j < 8; ++j) {
    const int col = w * 128 + j * 16 + (lane & 15);
    const float gm = gamma[col], bt = beta[col];
    for (int i = 0; i < 8; ++i) {
      float hn = (acc[j][i] - ldsMu[rloc + i]) * ldsRs[rloc + i] * gm + bt;
      hn_bf[(size_t)(rowB + i) * Hh + col] = (__bf16)hn;
    }
  }
}

// ---------------- K6: out = hn Wout^T + bout ----------------
// M=16384, N=1024, K=1024 — the dominant GEMM (34 GFLOP).
// Same async double-buffered 32x64 A staging as K1.
__global__ __launch_bounds__(256) void k_gemm_out(const __bf16* __restrict__ hn_bf,
                                                  const __bf16* __restrict__ Wout_bf,
                                                  const float* __restrict__ bout,
                                                  float* __restrict__ out) {
  __shared__ __bf16 ldsA[2][32 * 64];
  const int tid = threadIdx.x, lane = tid & 31, w = tid >> 5;
  const int wrow = w >> 2, wcol = w & 3;
  const int rowBlk = blockIdx.x * 32;
  const int n0 = blockIdx.y * 64 + wcol * 16;
  const int sr = tid >> 3, sc = (tid & 7) * 8;
  const __bf16* gA = hn_bf + (size_t)(rowBlk + sr) * Hh + sc;
  const int lofs = sr * 64 + sc;

  stage16B(&ldsA[0][lofs], gA);
  v8f acc = {};
  const int NI = Hh / 64;
  for (int i = 0; i < NI; ++i) {
    const int kk = i * 64;
    const int p = i & 1;
    if (i + 1 < NI) {
      stage16B(&ldsA[p ^ 1][lofs], gA + kk + 64);
      wait_async<1>();
    } else {
      wait_async<0>();
    }
    __syncthreads();
    const __bf16* bufA = &ldsA[p][0];
    for (int ks = 0; ks < 64; ks += 32) {
      v16bf a = frag_a_lds(bufA, 64, lane, wrow * 16, ks);
      v16bf b = frag_b_glb(Wout_bf, Hh, n0, kk + ks, lane);
      acc = wmma_bf16(a, b, acc);
    }
    __syncthreads();
  }
  const int col = n0 + (lane & 15);
  const int rowBase = rowBlk + wrow * 16 + ((lane >> 4) << 3);
  const float bo = bout[col];
  for (int i = 0; i < 8; ++i)
    out[(size_t)(rowBase + i) * Hh + col] = acc[i] + bo;
}

// ---------------- launcher ----------------
extern "C" void kernel_launch(void* const* d_in, const int* in_sizes, int n_in,
                              void* d_out, int out_size, void* d_ws, size_t ws_size,
                              hipStream_t stream) {
  const float* x     = (const float*)d_in[0];
  const float* A_log = (const float*)d_in[1];
  const float* WB    = (const float*)d_in[2];
  const float* WC    = (const float*)d_in[3];
  const float* Dv    = (const float*)d_in[4];
  const float* Wg    = (const float*)d_in[5];
  const float* bg    = (const float*)d_in[6];
  const float* Wout  = (const float*)d_in[7];
  const float* bout  = (const float*)d_in[8];
  const float* gamma = (const float*)d_in[9];
  const float* beta  = (const float*)d_in[10];

  float* out    = (float*)d_out;
  float* fs_out = out + (size_t)Mm * Hh;   // final_state appended after [B,T,H]

  // workspace carve (all region sizes 256B-aligned)
  char* p = (char*)d_ws;
  __bf16* xb      = (__bf16*)p; p += (size_t)Mm * Hh * 2;   // 32 MB
  __bf16* hn_bf   = (__bf16*)p; p += (size_t)Mm * Hh * 2;   // 32 MB
  __bf16* Wg_bf   = (__bf16*)p; p += (size_t)Ss * Hh * 2;
  __bf16* WB_bf   = (__bf16*)p; p += (size_t)Ss * Hh * 2;
  __bf16* WC_bf   = (__bf16*)p; p += (size_t)Hh * Ss * 2;
  __bf16* Wout_bf = (__bf16*)p; p += (size_t)Hh * Hh * 2;   // 2 MB
  float*  u       = (float*)p;  p += (size_t)Mm * Ss * 4;   // 4 MB
  float*  carry   = (float*)p;  p += (size_t)Bb * NC * Ss * 4;
  float*  startb  = (float*)p;  p += (size_t)Bb * NC * Ss * 4;
  __bf16* st_bf   = (__bf16*)p; p += (size_t)Mm * Ss * 2;   // 2 MB

  // 0) precision conversion to bf16 for WMMA operands
  k_cvt_bf16<<<(Mm * Hh + 255) / 256, 256, 0, stream>>>(x, xb, Mm * Hh);
  k_cvt_bf16<<<(Ss * Hh + 255) / 256, 256, 0, stream>>>(Wg, Wg_bf, Ss * Hh);
  k_cvt_bf16<<<(Ss * Hh + 255) / 256, 256, 0, stream>>>(WB, WB_bf, Ss * Hh);
  k_cvt_bf16<<<(Hh * Ss + 255) / 256, 256, 0, stream>>>(WC, WC_bf, Hh * Ss);
  k_cvt_bf16<<<(Hh * Hh + 255) / 256, 256, 0, stream>>>(Wout, Wout_bf, Hh * Hh);

  // 1) u = sigmoid(x Wg^T + bg) * (x WB^T)   [WMMA, async LDS staging]
  k_gemm_u<<<Mm / 32, 256, 0, stream>>>(xb, Wg_bf, WB_bf, bg, u);

  // 2-4) chunked linear recurrence scan over T
  k_scan_carry<<<Bb * NC, 64, 0, stream>>>(u, A_log, carry);
  k_scan_combine<<<1, 256, 0, stream>>>(carry, A_log, startb, fs_out);
  k_scan_states<<<Bb * NC, 64, 0, stream>>>(u, A_log, startb, st_bf);

  // 5) y = states WC^T + D*x + x, LayerNorm -> hn   [WMMA + LDS reduction]
  k_gemm_y_ln<<<Mm / 16, 256, 0, stream>>>(st_bf, WC_bf, x, Dv, gamma, beta, hn_bf);

  // 6) out = hn Wout^T + bout   [dominant WMMA GEMM, async LDS staging]
  {
    dim3 g(Mm / 32, Hh / 64);
    k_gemm_out<<<g, 256, 0, stream>>>(hn_bf, Wout_bf, bout, out);
  }
}
